// JanossyPooling_1408749273395
// MI455X (gfx1250) — compile-verified, matrix operand order checked by reading, not run on hardware
//
#include <hip/hip_runtime.h>

typedef _Float16 v16h __attribute__((ext_vector_type(16)));
typedef _Float16 v8h  __attribute__((ext_vector_type(8)));
typedef _Float16 v4h  __attribute__((ext_vector_type(4)));
typedef float    v8f  __attribute__((ext_vector_type(8)));

#define TM      64      // rows per block
#define THREADS 128     // 4 wave32s; wave w owns row-tile w (16 rows)

// Load one 16x32 f16 WMMA fragment half-row for this lane.
// Per ISA layout: lane holds 8 halves at k=+0 and 8 halves at k=+16.
__device__ __forceinline__ v16h load_frag16(const _Float16* p) {
  v8h lo = *(const v8h*)(p);
  v8h hi = *(const v8h*)(p + 16);
  return __builtin_shufflevector(lo, hi,
      0, 1, 2, 3, 4, 5, 6, 7, 8, 9, 10, 11, 12, 13, 14, 15);
}

// One wave computes a 16-row x 128-col layer for BOTH streams at once:
//   outF = relu(aF[16 x DIN] @ Wt^T + b),  outR = relu(aR[16 x DIN] @ Wt^T + b)
// Fusing the streams means each B fragment is loaded ONCE and fed to two
// WMMAs (halves the dominant L2 weight traffic).
// aF/aR: LDS activations, row-major [rows][aStride] f16 (aR ignored if !DUAL)
// Wt:    global f16 weights, transposed: Wt[n*DIN + k] = W[k][n]
// outF/outR: LDS f16 output buffers, fixed stride 136
// revR:  mirror 128-wide K-blocks of aR (reversed-concat stream, layer 0)
template<int DIN, bool DUAL>
__device__ __forceinline__ void gemm_tile(
    const _Float16* __restrict__ aF,
    const _Float16* __restrict__ aR, int aStride,
    const _Float16* __restrict__ Wt,
    const float* __restrict__ bias,
    _Float16* __restrict__ outF,
    _Float16* __restrict__ outR,
    int lane, int rowTile, int nBlk, bool revR)
{
  v8f accF[8], accR[8];
#pragma unroll
  for (int i = 0; i < 8; ++i)
#pragma unroll
    for (int e = 0; e < 8; ++e) {
      accF[i][e] = 0.0f;
      if (DUAL) accR[i][e] = 0.0f;
    }

  const int m    = lane & 15;           // A row / B col within tile
  const int koff = (lane >> 4) << 3;    // lanes 16-31 start at k=8
  const _Float16* rowF = aF + (rowTile * 16 + m) * aStride;
  const _Float16* rowR = DUAL ? (aR + (rowTile * 16 + m) * aStride) : aF;

#pragma unroll 2
  for (int k0 = 0; k0 < DIN; k0 += 32) {
    const v16h af = load_frag16(rowF + k0 + koff);
    v16h ar;
    if (DUAL) {
      int kA = k0;
      if (revR) {                        // mirrored 128-wide block order
        const int blk = k0 >> 7;
        kA = ((nBlk - 1 - blk) << 7) | (k0 & 127);
      }
      ar = load_frag16(rowR + kA + koff);
    }
#pragma unroll
    for (int ct = 0; ct < 8; ++ct) {
      const v16h b = load_frag16(Wt + (ct * 16 + m) * DIN + k0 + koff);
      accF[ct] = __builtin_amdgcn_wmma_f32_16x16x32_f16(
          false, af, false, b, (short)0, accF[ct], false, false);
      if (DUAL)
        accR[ct] = __builtin_amdgcn_wmma_f32_16x16x32_f16(
            false, ar, false, b, (short)0, accR[ct], false, false);
    }
  }

  // epilogue: bias + relu, f32 -> f16 into LDS (stride 136)
  const int mbase = rowTile * 16 + ((lane >> 4) << 3);  // lanes>=16 -> M=8+r
#pragma unroll
  for (int ct = 0; ct < 8; ++ct) {
    const int n = ct * 16 + m;
    const float bn = bias[n];
#pragma unroll
    for (int r = 0; r < 8; ++r) {
      float vf = accF[ct][r] + bn;
      vf = vf > 0.0f ? vf : 0.0f;
      outF[(mbase + r) * 136 + n] = (_Float16)vf;
      if (DUAL) {
        float vr = accR[ct][r] + bn;
        vr = vr > 0.0f ? vr : 0.0f;
        outR[(mbase + r) * 136 + n] = (_Float16)vr;
      }
    }
  }
}

// Fused level kernel: gather -> (fwd[+rev]) 3-layer MLP -> NH heads (D->1)
template<int L, int NH>
__global__ __launch_bounds__(THREADS)
void level_kernel(const float* __restrict__ h, const int* __restrict__ idx,
                  const _Float16* __restrict__ Wt0,
                  const _Float16* __restrict__ Wt1,
                  const _Float16* __restrict__ Wt2,
                  const float* __restrict__ b0,
                  const float* __restrict__ b1,
                  const float* __restrict__ b2,
                  const float* hw0, const float* hb0,
                  const float* hw1, const float* hb1,
                  const float* hw2, const float* hb2,
                  float* out0, float* out1, float* out2,
                  int N)
{
  extern __shared__ _Float16 smem[];
  const int tid        = threadIdx.x;
  const int lane       = tid & 31;
  const int wave       = tid >> 5;
  const int blockStart = blockIdx.x * TM;
  const int stride0    = L * 128 + 8;   // +8 halves: 16B pad vs bank conflicts

  _Float16* act0 = smem;
  _Float16* sFa  = act0 + TM * stride0;
  _Float16* sFb  = sFa + TM * 136;
  _Float16* sRa  = sFb + TM * 136;   // only sized/used when L > 1
  _Float16* sRb  = sRa + TM * 136;

  // ---- gather h rows (via idx for L>1) + fp32->fp16 convert into LDS ----
  const int totalChunks = TM * L * 32;       // float4 chunks
  for (int c = tid; c < totalChunks; c += THREADS) {
    const int slot = c >> 5;
    const int off  = (c & 31) << 2;
    const int i    = slot / L;
    const int j    = slot - i * L;
    int gr = blockStart + i;
    if (gr >= N) gr = N - 1;               // clamp tail (padding rows harmless)
    int src;
    if (L == 1) src = gr;
    else        src = idx[gr * L + j];
    const float4 v = *(const float4*)(h + (size_t)src * 128 + off);
    v4h hv;
    hv[0] = (_Float16)v.x; hv[1] = (_Float16)v.y;
    hv[2] = (_Float16)v.z; hv[3] = (_Float16)v.w;
    *(v4h*)(act0 + i * stride0 + j * 128 + off) = hv;
  }
  __syncthreads();

  constexpr bool DUAL = (L > 1);
  // ---- layer 0: [TM x L*128] @ [L*128 x 128], F + mirrored-R fused ----
  gemm_tile<L * 128, DUAL>(act0, act0, stride0, Wt0, b0, sFa, sRa,
                           lane, wave, L, true);
  __syncthreads();
  // ---- layer 1 ----
  gemm_tile<128, DUAL>(sFa, sRa, 136, Wt1, b1, sFb, sRb,
                       lane, wave, 1, false);
  __syncthreads();
  // ---- layer 2 (writes back into sFa / sRa) ----
  gemm_tile<128, DUAL>(sFb, sRb, 136, Wt2, b2, sFa, sRa,
                       lane, wave, 1, false);
  __syncthreads();

  // ---- heads: (relu3F + relu3R) @ W_head + b_head ----
  const float* HW[3]  = {hw0, hw1, hw2};
  const float* HB[3]  = {hb0, hb1, hb2};
  float*       OUT[3] = {out0, out1, out2};
  const int i  = tid & (TM - 1);
  const int gr = blockStart + i;
  for (int hid = tid >> 6; hid < NH; hid += 2) {
    const float* w = HW[hid];
    float s = HB[hid][0];
    for (int k = 0; k < 128; ++k) {
      float yv = (float)sFa[i * 136 + k];
      if (L > 1) yv += (float)sRa[i * 136 + k];
      s += yv * w[k];
    }
    if (gr < N) OUT[hid][gr] = s;
  }
}

// Transpose + convert weights to f16: Wt[n*Din + k] = (f16) W[k*128 + n]
__global__ void pack_weight(const float* __restrict__ W,
                            _Float16* __restrict__ Wt, int Din)
{
  const int e = blockIdx.x * blockDim.x + threadIdx.x;
  if (e < Din * 128) {
    const int k = e >> 7;
    const int n = e & 127;
    Wt[(size_t)n * Din + k] = (_Float16)W[(size_t)k * 128 + n];
  }
}

extern "C" void kernel_launch(void* const* d_in, const int* in_sizes, int n_in,
                              void* d_out, int out_size, void* d_ws, size_t ws_size,
                              hipStream_t stream) {
  const float* h    = (const float*)d_in[0];
  const int*   idx2 = (const int*)d_in[1];
  const int*   idx3 = (const int*)d_in[2];
  const int*   idx4 = (const int*)d_in[3];
  const int N1 = in_sizes[0] / 128;
  const int N2 = in_sizes[1] / 2;
  const int N3 = in_sizes[2] / 3;
  const int N4 = in_sizes[3] / 4;

  // ---- pack all MLP weights (f32 -> f16 transposed) into workspace ----
  _Float16* ws = (_Float16*)d_ws;
  size_t woff[4][3];
  size_t off = 0;
  for (int L = 1; L <= 4; ++L) {
    const int dins[3] = {L * 128, 128, 128};
    for (int l = 0; l < 3; ++l) {
      woff[L - 1][l] = off;
      const int inIdx = 4 + (L - 1) * 6 + l * 2;     // W0,b0,W1,b1,W2,b2
      const float* W = (const float*)d_in[inIdx];
      const int total = dins[l] * 128;
      pack_weight<<<(total + 255) / 256, 256, 0, stream>>>(W, ws + off, dins[l]);
      off += (size_t)total;
    }
  }

  float* out = (float*)d_out;
  const size_t o1 = 0;
  const size_t o2 = 3 * (size_t)N1;
  const size_t o3 = o2 + 2 * (size_t)N2;
  const size_t o4 = o3 + 2 * (size_t)N3;

  // ---- level 1: MLP(h) -> sigma, epsilon, q ----
  {
    const size_t shmem = 2 * ((size_t)TM * (128 + 8) + 2 * (size_t)TM * 136);
    hipFuncSetAttribute(reinterpret_cast<const void*>(&level_kernel<1, 3>),
                        hipFuncAttributeMaxDynamicSharedMemorySize, (int)shmem);
    level_kernel<1, 3><<<(N1 + TM - 1) / TM, THREADS, shmem, stream>>>(
        h, nullptr,
        ws + woff[0][0], ws + woff[0][1], ws + woff[0][2],
        (const float*)d_in[5], (const float*)d_in[7], (const float*)d_in[9],
        (const float*)d_in[28], (const float*)d_in[29],
        (const float*)d_in[30], (const float*)d_in[31],
        (const float*)d_in[32], (const float*)d_in[33],
        out + o1, out + o1 + N1, out + o1 + 2 * (size_t)N1, N1);
  }
  // ---- level 2 -> k2, eq2 ----
  {
    const size_t shmem = 2 * ((size_t)TM * (256 + 8) + 4 * (size_t)TM * 136);
    hipFuncSetAttribute(reinterpret_cast<const void*>(&level_kernel<2, 2>),
                        hipFuncAttributeMaxDynamicSharedMemorySize, (int)shmem);
    level_kernel<2, 2><<<(N2 + TM - 1) / TM, THREADS, shmem, stream>>>(
        h, idx2,
        ws + woff[1][0], ws + woff[1][1], ws + woff[1][2],
        (const float*)d_in[11], (const float*)d_in[13], (const float*)d_in[15],
        (const float*)d_in[34], (const float*)d_in[35],
        (const float*)d_in[36], (const float*)d_in[37],
        nullptr, nullptr,
        out + o2, out + o2 + N2, nullptr, N2);
  }
  // ---- level 3 -> k3, eq3 ----
  {
    const size_t shmem = 2 * ((size_t)TM * (384 + 8) + 4 * (size_t)TM * 136);
    hipFuncSetAttribute(reinterpret_cast<const void*>(&level_kernel<3, 2>),
                        hipFuncAttributeMaxDynamicSharedMemorySize, (int)shmem);
    level_kernel<3, 2><<<(N3 + TM - 1) / TM, THREADS, shmem, stream>>>(
        h, idx3,
        ws + woff[2][0], ws + woff[2][1], ws + woff[2][2],
        (const float*)d_in[17], (const float*)d_in[19], (const float*)d_in[21],
        (const float*)d_in[38], (const float*)d_in[39],
        (const float*)d_in[40], (const float*)d_in[41],
        nullptr, nullptr,
        out + o3, out + o3 + N3, nullptr, N3);
  }
  // ---- level 4 -> k4, eq4 ----
  {
    const size_t shmem = 2 * ((size_t)TM * (512 + 8) + 4 * (size_t)TM * 136);
    hipFuncSetAttribute(reinterpret_cast<const void*>(&level_kernel<4, 2>),
                        hipFuncAttributeMaxDynamicSharedMemorySize, (int)shmem);
    level_kernel<4, 2><<<(N4 + TM - 1) / TM, THREADS, shmem, stream>>>(
        h, idx4,
        ws + woff[3][0], ws + woff[3][1], ws + woff[3][2],
        (const float*)d_in[23], (const float*)d_in[25], (const float*)d_in[27],
        (const float*)d_in[42], (const float*)d_in[43],
        (const float*)d_in[44], (const float*)d_in[45],
        nullptr, nullptr,
        out + o4, out + o4 + N4, nullptr, N4);
  }
}
